// FlaxModel_68942815035401
// MI455X (gfx1250) — matrix operand order repaired and checked by reading, not compiled, not run on hardware
//
#include <hip/hip_runtime.h>
#include <hip/hip_bf16.h>

typedef __attribute__((ext_vector_type(16))) _Float16 v16h;
typedef __attribute__((ext_vector_type(8)))  _Float16 h8;
typedef __attribute__((ext_vector_type(8)))  float    v8f;

#define NL   2
#define RC_F 2.0f
#define PI_F 3.14159265358979f
#define EPS_F 1e-9f
#define WAVES 4

// ---- B-fragment arena layout (each fragment: 32 lanes x 16 halves = 1KB) ----
// W1 : frags 0..3   (kstep=0, ntile 0..3)           K 12->32, N=64
// W2 : frags 4..9   (kstep*3 + ntile)               K=64,     N=48
// Wv : frags 10..11 (kstep)                         K 48->64, N=16
// layer l block at 12 + 18*l:
//   +0..7  W_mlp1 (kstep*4 + ntile)                 K=64, N=64
//   +8..13 W_mlp2 (kstep*3 + ntile)                 K=64, N=48
//   +14,15 W_gate (kstep)                           K 48->64, N=16
//   +16,17 W_new  (kstep)                           K 48->64, N=16
#define FR_W1 0
#define FR_W2 4
#define FR_WV 10
#define FR_L0 12
#define FR_LSTRIDE 18
#define NFRAGS 48

struct __align__(64) SharedMem {
    _Float16 frags[NFRAGS * 512];   // 48 KB, pre-swizzled B fragments
    float b1[64];
    float b2[48];
    float bm1[NL][64];
    float bm2[NL][48];
    float wout[48];
    _Float16 buf[WAVES][16 * 64];   // per-wave A-matrix scratch (16 edges x 64)
};

// ---------------- WMMA helper ----------------
__device__ __forceinline__ v8f wmma16(v16h a, v16h b, v8f c) {
    return __builtin_amdgcn_wmma_f32_16x16x32_f16(
        false, a, false, b, (short)0, c, false, false);
}

// A fragment: buf row-major [16 rows][64 f16]. 16-bit A 16x32 layout:
// lanes 0-15 : row=lane,   K = {ks*32 + 0..7,  ks*32 + 16..23}
// lanes 16-31: row=lane-16,K = {ks*32 + 8..15, ks*32 + 24..31}
__device__ __forceinline__ v16h load_A(const _Float16* buf, int lane, int kstep) {
    int row  = lane & 15;
    int koff = (lane >> 4) * 8;
    const _Float16* p = buf + row * 64 + kstep * 32 + koff;
    h8 lo = *(const h8*)(p);
    h8 hi = *(const h8*)(p + 16);
    return __builtin_shufflevector(lo, hi, 0, 1, 2, 3, 4, 5, 6, 7,
                                           8, 9, 10, 11, 12, 13, 14, 15);
}

// B fragment: one contiguous 32-byte read per lane.
__device__ __forceinline__ v16h load_Bf(const _Float16* frags, int fragIdx, int lane) {
    return *(const v16h*)(frags + fragIdx * 512 + lane * 16);
}

// Stage a weight matrix into pre-swizzled B-fragment form.
// 16-bit B 32x16 layout: lane l -> col n = ntile*16 + (l&15),
//                        K = kstep*32 + (j>>3)*16 + (l>>4)*8 + (j&7), j=0..15.
__device__ __forceinline__ void stage_B(_Float16* dst, const float* W, int Kreal,
                                        int N, int ntiles, int ksteps,
                                        int tid, int nthr) {
    const int total = ksteps * ntiles * 512;
    for (int i = tid; i < total; i += nthr) {
        int frag  = i >> 9;
        int kstep = frag / ntiles;
        int ntile = frag - kstep * ntiles;
        int r = i & 511;
        int l = r >> 4;
        int j = r & 15;
        int n = ntile * 16 + (l & 15);
        int k = kstep * 32 + (j >> 3) * 16 + (l >> 4) * 8 + (j & 7);
        float v = (k < Kreal) ? W[k * N + n] : 0.0f;
        dst[i] = (_Float16)v;
    }
}

// Fast SiLU: v * rcp(1 + exp(-v)) — v_exp_f32 + v_rcp_f32, no IEEE divide.
__device__ __forceinline__ float silu(float v) {
    return v * __builtin_amdgcn_rcpf(1.0f + __expf(-v));
}

// ---------------- main kernel ----------------
__global__ void __launch_bounds__(WAVES * 32)
allegro_edge_kernel(const float* __restrict__ positions,
                    const int*   __restrict__ species,
                    const int*   __restrict__ senders,
                    const int*   __restrict__ receivers,
                    const float* __restrict__ W1g,  const float* __restrict__ b1g,
                    const float* __restrict__ W2g,  const float* __restrict__ b2g,
                    const float* __restrict__ Wvg,
                    const float* __restrict__ Wm1g, const float* __restrict__ bm1g,
                    const float* __restrict__ Wm2g, const float* __restrict__ bm2g,
                    const float* __restrict__ Wgg,  const float* __restrict__ Wng,
                    const float* __restrict__ Woutg,
                    float* __restrict__ out,
                    int Nnodes, int E) {
    __shared__ SharedMem sm;

    const int tid  = threadIdx.x;
    const int nthr = blockDim.x;

    // ---- one-time per-block: stage all weights as swizzled B fragments ----
    stage_B(sm.frags + FR_W1 * 512, W1g, 12, 64, 4, 1, tid, nthr);
    stage_B(sm.frags + FR_W2 * 512, W2g, 64, 48, 3, 2, tid, nthr);
    stage_B(sm.frags + FR_WV * 512, Wvg, 48, 16, 1, 2, tid, nthr);
#pragma unroll
    for (int l = 0; l < NL; ++l) {
        const int base = FR_L0 + FR_LSTRIDE * l;
        stage_B(sm.frags + (base + 0) * 512,  Wm1g + l * 64 * 64, 64, 64, 4, 2, tid, nthr);
        stage_B(sm.frags + (base + 8) * 512,  Wm2g + l * 64 * 48, 64, 48, 3, 2, tid, nthr);
        stage_B(sm.frags + (base + 14) * 512, Wgg  + l * 48 * 16, 48, 16, 1, 2, tid, nthr);
        stage_B(sm.frags + (base + 16) * 512, Wng  + l * 48 * 16, 48, 16, 1, 2, tid, nthr);
    }
    for (int i = tid; i < 64; i += nthr) sm.b1[i] = b1g[i];
    for (int i = tid; i < 48; i += nthr) { sm.b2[i] = b2g[i]; sm.wout[i] = Woutg[i]; }
    for (int i = tid; i < NL * 64; i += nthr) (&sm.bm1[0][0])[i] = bm1g[i];
    for (int i = tid; i < NL * 48; i += nthr) (&sm.bm2[0][0])[i] = bm2g[i];
    __syncthreads();

    const int wave = tid >> 5;
    const int lane = tid & 31;
    const int hi   = lane >> 4;
    const int nlo  = lane & 15;
    _Float16* buf  = sm.buf[wave];
    const _Float16* frags = sm.frags;

    const int tilesTotal = (E + 15) / 16;

    for (int tile = blockIdx.x * WAVES + wave; tile < tilesTotal;
         tile += gridDim.x * WAVES) {
        const int e0 = tile * 16;

        // ---------- prologue: per-edge scalars ----------
        const int  myE     = e0 + nlo;
        const bool myValid = (myE < E);
        const int  sIdx    = myValid ? senders[myE]   : 0;
        const int  rIdx    = myValid ? receivers[myE] : 0;

        float rx = positions[rIdx * 3 + 0] - positions[sIdx * 3 + 0];
        float ry = positions[rIdx * 3 + 1] - positions[sIdx * 3 + 1];
        float rz = positions[rIdx * 3 + 2] - positions[sIdx * 3 + 2];
        float d    = sqrtf(rx * rx + ry * ry + rz * rz);
        float dinv = __builtin_amdgcn_rcpf(d + EPS_F);
        float ux = rx * dinv, uy = ry * dinv, uz = rz * dinv;

        float vm = myValid ? 1.0f : 0.0f;
        float tc = fminf(fmaxf(d * (1.0f / RC_F), 0.0f), 1.0f);
        float cut = 0.5f * (__cosf(PI_F * tc) + 1.0f) * vm;

        if (lane < 16) {
            const int sp_s = species[sIdx];
            const int sp_r = species[rIdx];
            _Float16* p = buf + nlo * 64;
            p[0] = (_Float16)(vm * (sp_s == 0 ? 1.0f : 0.0f));
            p[1] = (_Float16)(vm * (sp_s == 1 ? 1.0f : 0.0f));
            p[2] = (_Float16)(vm * (sp_r == 0 ? 1.0f : 0.0f));
            p[3] = (_Float16)(vm * (sp_r == 1 ? 1.0f : 0.0f));
            // Bessel: sin(n*pi*d/RC)/(d+eps) * sqrt(2/RC); RC=2 -> factor 1
#pragma unroll
            for (int nb = 1; nb <= 8; ++nb) {
                float bes = __sinf((float)nb * (PI_F / RC_F) * d) * dinv;
                p[3 + nb] = (_Float16)(vm * bes);
            }
#pragma unroll
            for (int j = 12; j < 32; ++j) p[j] = (_Float16)0.0f;
        }

        // per-edge scalars into C/D row layout: row r -> edge r + 8*hi
        float cutv[8], uxv[8], uyv[8], uzv[8];
        int   rcvv[8];
#pragma unroll
        for (int r = 0; r < 8; ++r) {
            int src = r + 8 * hi;
            cutv[r] = __shfl(cut, src);
            uxv[r]  = __shfl(ux, src);
            uyv[r]  = __shfl(uy, src);
            uzv[r]  = __shfl(uz, src);
            rcvv[r] = __shfl(rIdx, src);
        }

        // ---------- H1 = silu(feat @ W1 + b1)  [16x64], K=32 ----------
        {
            v16h aF = load_A(buf, lane, 0);
#pragma unroll
            for (int nt = 0; nt < 4; ++nt) {
                v8f acc = {};
                acc = wmma16(aF, load_Bf(frags, FR_W1 + nt, lane), acc);
                float bias = sm.b1[nt * 16 + nlo];
#pragma unroll
                for (int r = 0; r < 8; ++r) {
                    float v = silu(acc[r] + bias);
                    buf[(r + 8 * hi) * 64 + nt * 16 + nlo] = (_Float16)v;
                }
            }
        }

        // ---------- x = (H1 @ W2 + b2) * cut  [16x48], K=64 ----------
        float xf[3][8];
        {
            v16h a0 = load_A(buf, lane, 0);
            v16h a1 = load_A(buf, lane, 1);
#pragma unroll
            for (int nt = 0; nt < 3; ++nt) {
                v8f acc = {};
                acc = wmma16(a0, load_Bf(frags, FR_W2 + 0 * 3 + nt, lane), acc);
                acc = wmma16(a1, load_Bf(frags, FR_W2 + 1 * 3 + nt, lane), acc);
                float bias = sm.b2[nt * 16 + nlo];
#pragma unroll
                for (int r = 0; r < 8; ++r)
                    xf[nt][r] = (acc[r] + bias) * cutv[r];
            }
        }
#pragma unroll
        for (int nt = 0; nt < 3; ++nt)
#pragma unroll
            for (int r = 0; r < 8; ++r)
                buf[(r + 8 * hi) * 64 + nt * 16 + nlo] = (_Float16)xf[nt][r];

        // ---------- V init: V[i] = u_i * (x @ Wv) (Wv rows 48..63 zero) ----------
        float Vx[8], Vy[8], Vz[8];
        {
            v16h a0 = load_A(buf, lane, 0);
            v16h a1 = load_A(buf, lane, 1);
            v8f acc = {};
            acc = wmma16(a0, load_Bf(frags, FR_WV + 0, lane), acc);
            acc = wmma16(a1, load_Bf(frags, FR_WV + 1, lane), acc);
#pragma unroll
            for (int r = 0; r < 8; ++r) {
                float xw = acc[r];
                Vx[r] = uxv[r] * xw;
                Vy[r] = uyv[r] * xw;
                Vz[r] = uzv[r] * xw;
            }
        }

        // ---------- Allegro layers ----------
#pragma unroll
        for (int l = 0; l < NL; ++l) {
            const int base = FR_L0 + FR_LSTRIDE * l;
            // inv -> cols 48..63 (h = [x | inv], K=64)
#pragma unroll
            for (int r = 0; r < 8; ++r) {
                float inv = Vx[r] * Vx[r] + Vy[r] * Vy[r] + Vz[r] * Vz[r];
                buf[(r + 8 * hi) * 64 + 48 + nlo] = (_Float16)inv;
            }
            // T = silu(h @ W_mlp1 + b)
            {
                v16h a0 = load_A(buf, lane, 0);
                v16h a1 = load_A(buf, lane, 1);
#pragma unroll
                for (int nt = 0; nt < 4; ++nt) {
                    v8f acc = {};
                    acc = wmma16(a0, load_Bf(frags, base + 0 * 4 + nt, lane), acc);
                    acc = wmma16(a1, load_Bf(frags, base + 1 * 4 + nt, lane), acc);
                    float bias = sm.bm1[l][nt * 16 + nlo];
#pragma unroll
                    for (int r = 0; r < 8; ++r) {
                        float v = silu(acc[r] + bias);
                        buf[(r + 8 * hi) * 64 + nt * 16 + nlo] = (_Float16)v;
                    }
                }
            }
            // dx = T @ W_mlp2 + b ; x = (x + dx) * cut
            {
                v16h a0 = load_A(buf, lane, 0);
                v16h a1 = load_A(buf, lane, 1);
#pragma unroll
                for (int nt = 0; nt < 3; ++nt) {
                    v8f acc = {};
                    acc = wmma16(a0, load_Bf(frags, base + 8 + 0 * 3 + nt, lane), acc);
                    acc = wmma16(a1, load_Bf(frags, base + 8 + 1 * 3 + nt, lane), acc);
                    float bias = sm.bm2[l][nt * 16 + nlo];
#pragma unroll
                    for (int r = 0; r < 8; ++r)
                        xf[nt][r] = (xf[nt][r] + acc[r] + bias) * cutv[r];
                }
            }
            // refresh x in buf cols 0..47 (stale cols 48..63 hit zero B-rows)
#pragma unroll
            for (int nt = 0; nt < 3; ++nt)
#pragma unroll
                for (int r = 0; r < 8; ++r)
                    buf[(r + 8 * hi) * 64 + nt * 16 + nlo] = (_Float16)xf[nt][r];

            // gate / new projections, V update
            {
                v16h a0 = load_A(buf, lane, 0);
                v16h a1 = load_A(buf, lane, 1);
                v8f g = {};
                g = wmma16(a0, load_Bf(frags, base + 14, lane), g);
                g = wmma16(a1, load_Bf(frags, base + 15, lane), g);
                v8f w = {};
                w = wmma16(a0, load_Bf(frags, base + 16, lane), w);
                w = wmma16(a1, load_Bf(frags, base + 17, lane), w);
#pragma unroll
                for (int r = 0; r < 8; ++r) {
                    Vx[r] = Vx[r] * g[r] + uxv[r] * w[r];
                    Vy[r] = Vy[r] * g[r] + uyv[r] * w[r];
                    Vz[r] = Vz[r] * g[r] + uzv[r] * w[r];
                }
            }
        }

        // ---------- readout + scatter ----------
        float pr[8];
#pragma unroll
        for (int r = 0; r < 8; ++r)
            pr[r] = xf[0][r] * sm.wout[nlo] +
                    xf[1][r] * sm.wout[16 + nlo] +
                    xf[2][r] * sm.wout[32 + nlo];
#pragma unroll
        for (int m = 8; m >= 1; m >>= 1)
#pragma unroll
            for (int r = 0; r < 8; ++r)
                pr[r] += __shfl_xor(pr[r], m);

        if (nlo == 0) {
#pragma unroll
            for (int r = 0; r < 8; ++r) {
                int edge = e0 + r + 8 * hi;
                if (edge < E) {
                    float e = pr[r] * cutv[r] * (1.0f / 3.0f);  // / AVG_NEIGH
                    atomicAdd(&out[rcvv[r]], e);
                }
            }
        }
    }
}

// ---------------- launcher ----------------
extern "C" void kernel_launch(void* const* d_in, const int* in_sizes, int n_in,
                              void* d_out, int out_size, void* d_ws, size_t ws_size,
                              hipStream_t stream) {
    const float* positions = (const float*)d_in[0];
    const int*   species   = (const int*)d_in[1];
    const int*   senders   = (const int*)d_in[2];
    const int*   receivers = (const int*)d_in[3];
    const float* W1   = (const float*)d_in[4];
    const float* b1   = (const float*)d_in[5];
    const float* W2   = (const float*)d_in[6];
    const float* b2   = (const float*)d_in[7];
    const float* Wv   = (const float*)d_in[8];
    const float* Wm1  = (const float*)d_in[9];
    const float* bm1  = (const float*)d_in[10];
    const float* Wm2  = (const float*)d_in[11];
    const float* bm2  = (const float*)d_in[12];
    const float* Wg   = (const float*)d_in[13];
    const float* Wn   = (const float*)d_in[14];
    const float* Wout = (const float*)d_in[15];

    const int N = in_sizes[0] / 3;
    const int E = in_sizes[2];

    hipMemsetAsync(d_out, 0, (size_t)N * sizeof(float), stream);

    const int tiles = (E + 15) / 16;
    int blocks = (tiles + WAVES - 1) / WAVES;
    if (blocks > 4096) blocks = 4096;

    allegro_edge_kernel<<<blocks, WAVES * 32, 0, stream>>>(
        positions, species, senders, receivers,
        W1, b1, W2, b2, Wv, Wm1, bm1, Wm2, bm2, Wg, Wn, Wout,
        (float*)d_out, N, E);
}
